// GQA_61641370632474
// MI455X (gfx1250) — matrix operand order repaired and checked by reading, not compiled, not run on hardware
//
#include <hip/hip_runtime.h>
#include <hip/hip_bf16.h>
#include <math.h>

// ---------------------------------------------------------------------------
// GQA attention block for MI455X (gfx1250, wave32, WMMA 16x16x32 f16->f32)
// Pipeline: cast f32->f16, Q/K proj + RoPE, V proj (transposed), flash attn,
// output projection. Matmuls via v_wmma_f32_16x16x32_f16; activation tiles
// staged to LDS with async global->LDS copies (double buffered).
// ---------------------------------------------------------------------------

#define SEQ   2048
#define DM    4096
#define HD    128
#define NH    32
#define NKV   8

typedef _Float16 v16h __attribute__((ext_vector_type(16)));
typedef _Float16 v8h  __attribute__((ext_vector_type(8)));
typedef _Float16 v4h  __attribute__((ext_vector_type(4)));
typedef float    v8f  __attribute__((ext_vector_type(8)));
typedef float    v4f  __attribute__((ext_vector_type(4)));

union ABFrag { v16h full; v8h h8[2]; };

#if defined(__AMDGCN__) && __has_builtin(__builtin_amdgcn_global_load_async_to_lds_b128)
#define HAVE_ASYNC_LDS 1
#else
#define HAVE_ASYNC_LDS 0
#endif

#if HAVE_ASYNC_LDS
typedef int vi4 __attribute__((vector_size(16)));
typedef __attribute__((address_space(1))) vi4* gvi4_p;   // global (AS1), non-const
typedef __attribute__((address_space(3))) vi4* lvi4_p;   // LDS (AS3)
#endif

__device__ __forceinline__ v8f wmma_f16(v16h a, v16h b, v8f c) {
    // D = A(16x32) * B(32x16) + C, f32 accumulate
    return __builtin_amdgcn_wmma_f32_16x16x32_f16(
        false, a, false, b, (short)0, c, false, false);
}

// async global -> LDS copy of one 16B chunk (per lane), ASYNCcnt-tracked
__device__ __forceinline__ void async_copy_b128(const _Float16* g, _Float16* l) {
#if HAVE_ASYNC_LDS
    __builtin_amdgcn_global_load_async_to_lds_b128(
        (gvi4_p)(void*)g, (lvi4_p)(void*)l, 0, 0);
#else
    *(v8h*)l = *(const v8h*)g;   // synchronous fallback via VGPRs
#endif
}

__device__ __forceinline__ void wait_async0() {
#if HAVE_ASYNC_LDS
#if __has_builtin(__builtin_amdgcn_s_wait_asynccnt)
    __builtin_amdgcn_s_wait_asynccnt(0);
#else
    asm volatile("s_wait_asynccnt 0x0" ::: "memory");
#endif
#endif
}

// reductions within each 16-lane half-wave (a C-fragment row)
__device__ __forceinline__ float halfmax16(float v) {
#pragma unroll
    for (int m = 1; m < 16; m <<= 1) v = fmaxf(v, __shfl_xor(v, m, 32));
    return v;
}
__device__ __forceinline__ float halfsum16(float v) {
#pragma unroll
    for (int m = 1; m < 16; m <<= 1) v += __shfl_xor(v, m, 32);
    return v;
}

// ---------------------------------------------------------------------------
// elementwise f32 -> f16 cast (vectorized x4)
// ---------------------------------------------------------------------------
__global__ void cvt_f32_to_f16(const float* __restrict__ in,
                               _Float16* __restrict__ out, size_t n4) {
    size_t i = (size_t)blockIdx.x * blockDim.x + threadIdx.x;
    size_t stride = (size_t)gridDim.x * blockDim.x;
    for (; i < n4; i += stride) {
        v4f x = *(const v4f*)(in + 4 * i);
        v4h h;
        h.x = (_Float16)x.x; h.y = (_Float16)x.y;
        h.z = (_Float16)x.z; h.w = (_Float16)x.w;
        *(v4h*)(out + 4 * i) = h;
    }
}

// ---------------------------------------------------------------------------
// GEMM: C(M x N) = A(M x K) * W(N x K)^T, A/W in f16 row-major, f32 accum.
// Wave tile 32x64 (2 m-tiles x 4 n-tiles), block = 8 waves = 64m x 256n.
// A block (64 x 32 f16) staged to LDS via async copies, double buffered.
// Epilogues:
//   0: RoPE + store f16 as (head, seq, 128)          (Q and K projections)
//   1: store f16 transposed as (head, 128, seq)      (V projection)
//   2: store f32 row-major (M x N)                   (output projection)
// ---------------------------------------------------------------------------
#define EPI_ROPE 0
#define EPI_VT   1
#define EPI_F32  2

template <int EPI>
__global__ __launch_bounds__(256, 1)
void gemm_wmma(const _Float16* __restrict__ A, const _Float16* __restrict__ W,
               const float* __restrict__ rcos, const float* __restrict__ rsin,
               _Float16* __restrict__ outh, float* __restrict__ outf,
               int K, int N) {
    __shared__ _Float16 Abuf[2][64 * 32];   // 2 x 4KB double buffer

    const int tid  = threadIdx.x;
    const int wave = tid >> 5;
    const int lane = tid & 31;
    const int lo16 = lane & 15;
    const int hi   = lane >> 4;

    const int blockM = blockIdx.y * 64;
    const int m0 = blockM + (wave >> 2) * 32;
    const int n0 = blockIdx.x * 256 + (wave & 3) * 64;

    // cooperative A staging: 256 threads x 16B = 64 rows x 64B per k-step
    const int cr = tid >> 2;         // row within block tile (0..63)
    const int cc = (tid & 3) * 8;    // col chunk in halfs (0,8,16,24)
    const _Float16* agsrc = A + (size_t)(blockM + cr) * K + cc;
    const int lds_cpy = cr * 32 + cc;

    // per-wave A-fragment rows in LDS (halfs)
    const int arow0 = ((wave >> 2) * 32 + lo16) * 32;
    const int arow1 = ((wave >> 2) * 32 + 16 + lo16) * 32;

    const _Float16* brow[4];
#pragma unroll
    for (int j = 0; j < 4; ++j)
        brow[j] = W + (size_t)(n0 + 16 * j + lo16) * K;

    v8f acc[2][4];
#pragma unroll
    for (int mi = 0; mi < 2; ++mi)
#pragma unroll
        for (int j = 0; j < 4; ++j) acc[mi][j] = (v8f)0.0f;

    async_copy_b128(agsrc, &Abuf[0][lds_cpy]);

    for (int k0 = 0; k0 < K; k0 += 32) {
        const int cur = (k0 >> 5) & 1;
        wait_async0();          // this wave's staged chunks are in LDS
        __syncthreads();        // everyone's chunks are in LDS; prior reads done

        if (k0 + 32 < K)
            async_copy_b128(agsrc + k0 + 32, &Abuf[cur ^ 1][lds_cpy]);

        ABFrag a0, a1;
        a0.h8[0] = *(const v8h*)(&Abuf[cur][arow0 + 16 * hi]);
        a0.h8[1] = *(const v8h*)(&Abuf[cur][arow0 + 16 * hi + 8]);
        a1.h8[0] = *(const v8h*)(&Abuf[cur][arow1 + 16 * hi]);
        a1.h8[1] = *(const v8h*)(&Abuf[cur][arow1 + 16 * hi + 8]);

        v16h b[4];
#pragma unroll
        for (int j = 0; j < 4; ++j) {
            b[j] = *(const v16h*)(brow[j] + k0 + 16 * hi);
            __builtin_prefetch(brow[j] + k0 + 64, 0, 1);  // global_prefetch
        }
#pragma unroll
        for (int j = 0; j < 4; ++j) {
            acc[0][j] = wmma_f16(a0.full, b[j], acc[0][j]);
            acc[1][j] = wmma_f16(a1.full, b[j], acc[1][j]);
        }
    }

    // Epilogue. C layout: VGPR r holds row (r + 8*hi), column (n-tile + lo16).
#pragma unroll
    for (int mi = 0; mi < 2; ++mi) {
#pragma unroll
        for (int j = 0; j < 4; ++j) {
            const int nc = n0 + 16 * j + lo16;
#pragma unroll
            for (int r = 0; r < 8; ++r) {
                const int srow = m0 + mi * 16 + r + 8 * hi;
                float v = acc[mi][j][r];
                if constexpr (EPI == EPI_F32) {
                    outf[(size_t)srow * N + nc] = v;
                } else if constexpr (EPI == EPI_VT) {
                    // (head, dim, seq)
                    outh[((size_t)(nc >> 7) * HD + (nc & 127)) * SEQ + srow] =
                        (_Float16)v;
                } else {
                    // RoPE: pair (even,odd) along dim = adjacent lanes
                    float partner = __shfl_xor(v, 1, 32);
                    int i = (nc >> 1) & 63;
                    float c = rcos[(size_t)srow * 64 + i];
                    float s = rsin[(size_t)srow * 64 + i];
                    float res = (nc & 1) ? (partner * s + v * c)
                                         : (v * c - partner * s);
                    // (head, seq, 128)
                    outh[((size_t)(nc >> 7) * SEQ + srow) * HD + (nc & 127)] =
                        (_Float16)res;
                }
            }
        }
    }
}

// ---------------------------------------------------------------------------
// Flash attention, causal, GQA (4 Q heads per KV head).
// One wave per 16 query rows; block = 4 waves = 64 rows; grid (SEQ/64, NH).
// Q: (NH, SEQ, 128) f16, K: (NKV, SEQ, 128) f16, Vt: (NKV, 128, SEQ) f16.
// Output: attn16 (SEQ, NH*128) f16.
// ---------------------------------------------------------------------------
__global__ __launch_bounds__(128, 1)
void attn_flash(const _Float16* __restrict__ Qh, const _Float16* __restrict__ Kh,
                const _Float16* __restrict__ Vt, _Float16* __restrict__ attn16) {
    __shared__ _Float16 Pb[4][16 * 32];  // per-wave P tile (16 q x 32 k)

    const int wave = threadIdx.x >> 5;
    const int lane = threadIdx.x & 31;
    const int lo16 = lane & 15;
    const int hi   = lane >> 4;

    const int h   = blockIdx.y;
    const int hkv = h >> 2;
    const int qm0 = blockIdx.x * 64 + wave * 16;
    const float scale = 0.08838834764831845f;  // 1/sqrt(128)

    // Preload Q A-fragments for all four 32-wide d-chunks
    ABFrag qa[4];
    const _Float16* qrowp = Qh + ((size_t)h * SEQ + qm0 + lo16) * HD;
#pragma unroll
    for (int c = 0; c < 4; ++c) {
        qa[c].h8[0] = *(const v8h*)(qrowp + 32 * c + 16 * hi);
        qa[c].h8[1] = *(const v8h*)(qrowp + 32 * c + 16 * hi + 8);
    }

    v8f O[8];
#pragma unroll
    for (int t = 0; t < 8; ++t) O[t] = (v8f)0.0f;
    float rowM[8], rowL[8];
#pragma unroll
    for (int r = 0; r < 8; ++r) { rowM[r] = -3.0e38f; rowL[r] = 0.0f; }

    const _Float16* kbase = Kh + (size_t)hkv * SEQ * HD;
    const _Float16* vbase = Vt + (size_t)hkv * HD * SEQ;
    _Float16* pb = &Pb[wave][0];

    for (int k0 = 0; k0 <= qm0 + 15; k0 += 32) {
        // ---- S = Q K^T for 32 keys (two 16x16 tiles) ----
        v8f S0 = (v8f)0.0f, S1 = (v8f)0.0f;
#pragma unroll
        for (int c = 0; c < 4; ++c) {
            v16h b0 = *(const v16h*)(kbase + (size_t)(k0 + lo16) * HD +
                                     32 * c + 16 * hi);
            v16h b1 = *(const v16h*)(kbase + (size_t)(k0 + 16 + lo16) * HD +
                                     32 * c + 16 * hi);
            S0 = wmma_f16(qa[c].full, b0, S0);
            S1 = wmma_f16(qa[c].full, b1, S1);
        }

        // ---- online softmax (per row = per VGPR, across 16-lane half) ----
        float corr[8];
#pragma unroll
        for (int r = 0; r < 8; ++r) {
            const int qrow = qm0 + r + 8 * hi;
            float s0 = S0[r] * scale;
            float s1 = S1[r] * scale;
            s0 = (k0 + lo16 <= qrow) ? s0 : -3.0e38f;       // causal mask
            s1 = (k0 + 16 + lo16 <= qrow) ? s1 : -3.0e38f;
            float bm = halfmax16(fmaxf(s0, s1));
            float nm = fmaxf(rowM[r], bm);
            float c0 = __expf(rowM[r] - nm);
            float p0 = __expf(s0 - nm);
            float p1 = __expf(s1 - nm);
            rowL[r] = rowL[r] * c0 + halfsum16(p0 + p1);
            rowM[r] = nm;
            corr[r] = c0;
            // write P row-major into wave-private LDS: row (r+8*hi)
            const int ml = r + 8 * hi;
            pb[ml * 32 + lo16]      = (_Float16)p0;
            pb[ml * 32 + 16 + lo16] = (_Float16)p1;
        }

        // rescale O accumulators
#pragma unroll
        for (int t = 0; t < 8; ++t)
#pragma unroll
            for (int r = 0; r < 8; ++r) O[t][r] *= corr[r];

        // LDS store -> load in same wave: wait on DScnt
        asm volatile("s_wait_dscnt 0x0" ::: "memory");

        // read P back as an A-fragment (16x32, f16)
        ABFrag pa;
        pa.h8[0] = *(const v8h*)(pb + lo16 * 32 + 16 * hi);
        pa.h8[1] = *(const v8h*)(pb + lo16 * 32 + 16 * hi + 8);

        // ---- O += P V : 8 n-tiles of 16 dims ----
#pragma unroll
        for (int t = 0; t < 8; ++t) {
            v16h bv = *(const v16h*)(vbase + (size_t)(t * 16 + lo16) * SEQ +
                                     k0 + 16 * hi);
            O[t] = wmma_f16(pa.full, bv, O[t]);
        }
    }

    // ---- finalize: divide by row sums, store f16 (seq, NH*128) ----
    float inv[8];
#pragma unroll
    for (int r = 0; r < 8; ++r) inv[r] = 1.0f / rowL[r];
#pragma unroll
    for (int t = 0; t < 8; ++t) {
#pragma unroll
        for (int r = 0; r < 8; ++r) {
            const int srow = qm0 + r + 8 * hi;
            const int col  = h * HD + t * 16 + lo16;
            attn16[(size_t)srow * DM + col] = (_Float16)(O[t][r] * inv[r]);
        }
    }
}

// ---------------------------------------------------------------------------
// Host-side orchestration
// ---------------------------------------------------------------------------
extern "C" void kernel_launch(void* const* d_in, const int* in_sizes, int n_in,
                              void* d_out, int out_size, void* d_ws, size_t ws_size,
                              hipStream_t stream) {
    const float* x   = (const float*)d_in[0];
    const float* fct = (const float*)d_in[1];
    const float* fst = (const float*)d_in[2];
    const float* wq  = (const float*)d_in[3];
    const float* wk  = (const float*)d_in[4];
    const float* wv  = (const float*)d_in[5];
    const float* wo  = (const float*)d_in[6];
    float* out = (float*)d_out;

    _Float16* ws = (_Float16*)d_ws;
    size_t off = 0;
    _Float16* xh  = ws + off; off += (size_t)SEQ * DM;        // 8M
    _Float16* wqh = ws + off; off += (size_t)DM * DM;         // 16M
    _Float16* wkh = ws + off; off += (size_t)NKV * HD * DM;   // 4M
    _Float16* wvh = ws + off; off += (size_t)NKV * HD * DM;   // 4M
    _Float16* woh = ws + off; off += (size_t)DM * DM;         // 16M
    _Float16* Qh  = ws + off; off += (size_t)NH * SEQ * HD;   // 8M
    _Float16* Kh  = ws + off; off += (size_t)NKV * SEQ * HD;  // 2M
    _Float16* Vth = ws + off; off += (size_t)NKV * HD * SEQ;  // 2M
    _Float16* at  = ws + off; off += (size_t)SEQ * DM;        // 8M

    auto cvt = [&](const float* src, _Float16* dst, size_t n) {
        size_t n4 = n / 4;
        int blocks = (int)((n4 + 255) / 256);
        cvt_f32_to_f16<<<blocks, 256, 0, stream>>>(src, dst, n4);
    };
    cvt(x,  xh,  (size_t)SEQ * DM);
    cvt(wq, wqh, (size_t)DM * DM);
    cvt(wk, wkh, (size_t)NKV * HD * DM);
    cvt(wv, wvh, (size_t)NKV * HD * DM);
    cvt(wo, woh, (size_t)DM * DM);

    // Q = x @ wq^T + RoPE -> (NH, SEQ, 128)
    gemm_wmma<EPI_ROPE><<<dim3(DM / 256, SEQ / 64), 256, 0, stream>>>(
        xh, wqh, fct, fst, Qh, nullptr, DM, DM);
    // K = x @ wk^T + RoPE -> (NKV, SEQ, 128)
    gemm_wmma<EPI_ROPE><<<dim3((NKV * HD) / 256, SEQ / 64), 256, 0, stream>>>(
        xh, wkh, fct, fst, Kh, nullptr, DM, NKV * HD);
    // V = x @ wv^T -> transposed (NKV, 128, seq)
    gemm_wmma<EPI_VT><<<dim3((NKV * HD) / 256, SEQ / 64), 256, 0, stream>>>(
        xh, wvh, nullptr, nullptr, Vth, nullptr, DM, NKV * HD);

    // causal flash attention -> attn16 (SEQ, 4096)
    attn_flash<<<dim3(SEQ / 64, NH), 128, 0, stream>>>(Qh, Kh, Vth, at);

    // out = attn @ wo^T -> f32 (SEQ, 4096)
    gemm_wmma<EPI_F32><<<dim3(DM / 256, SEQ / 64), 256, 0, stream>>>(
        at, woh, nullptr, nullptr, nullptr, out, DM, DM);
}
